// GraphormerModel_71159018160675
// MI455X (gfx1250) — matrix operand order repaired and checked by reading, not compiled, not run on hardware
//
#include <hip/hip_runtime.h>
#include <hip/hip_bf16.h>
#include <math.h>

typedef _Float16 half_t;
typedef __attribute__((ext_vector_type(16))) _Float16 v16h;
typedef __attribute__((ext_vector_type(8)))  _Float16 v8h;
typedef __attribute__((ext_vector_type(8)))  float    v8f;

#define B_  16
#define N_  512
#define E_  4096
#define D_  256
#define H_  8
#define HD_ 32
#define L_  3
#define IN_ 768
#define FF_ 1024

// ---------------------------------------------------------------------------
// Device helpers
// ---------------------------------------------------------------------------
__device__ __forceinline__ float gelu_exact(float x) {
  return 0.5f * x * (1.0f + erff(x * 0.70710678118654752440f));
}

__device__ __forceinline__ v8f wmma_f16(v16h a, v16h b, v8f c) {
  return __builtin_amdgcn_wmma_f32_16x16x32_f16(
      /*neg_a=*/false, a, /*neg_b=*/false, b,
      /*c_mod=*/(short)0, c, /*reuse_a=*/false, /*reuse_b=*/false);
}

// A fragment (16x32, f16) from row-major A, tile origin (row0, k0).
// ISA 7.12.2: lane<16 row M=lane holds K {0..7,16..23}; lane>=16 holds {8..15,24..31}.
__device__ __forceinline__ v16h load_a_frag(const half_t* __restrict__ A, int lda,
                                            int row0, int k0, int lane) {
  int m  = lane & 15;
  int kb = (lane < 16) ? 0 : 8;
  const half_t* p = A + (size_t)(row0 + m) * lda + k0 + kb;
  v8h lo = *(const v8h*)(p);
  v8h hi = *(const v8h*)(p + 16);
  v16h a;
#pragma unroll
  for (int j = 0; j < 8; ++j) { a[j] = lo[j]; a[j + 8] = hi[j]; }
  return a;
}

// Same A-fragment gather but from LDS (f16 probs), row stride N_ halves.
__device__ __forceinline__ v16h load_a_frag_lds(const half_t* p16, int k0, int lane) {
  int m  = lane & 15;
  int kb = (lane < 16) ? 0 : 8;
  const half_t* p = p16 + m * N_ + k0 + kb;
  v8h lo = *(const v8h*)(p);        // ds_load_b128
  v8h hi = *(const v8h*)(p + 16);   // ds_load_b128
  v16h a;
#pragma unroll
  for (int j = 0; j < 8; ++j) { a[j] = lo[j]; a[j + 8] = hi[j]; }
  return a;
}

// B fragment (32x16) whose element b[kk][n] = M2[(col0+n)*ldm + k0+kk]
// i.e. M2 is the TRANSPOSE (N x K row-major) of the logical K x N matrix.
// Per lane: K {0..7,16..23} (lane<16) or {8..15,24..31} (lane>=16) -> two 16-byte vector loads.
__device__ __forceinline__ v16h load_b_frag_transposed(const half_t* __restrict__ M2, int ldm,
                                                       int col0, int k0, int lane) {
  int n  = lane & 15;
  int kb = (lane < 16) ? 0 : 8;
  const half_t* p = M2 + (size_t)(col0 + n) * ldm + k0 + kb;
  v8h lo = *(const v8h*)(p);
  v8h hi = *(const v8h*)(p + 16);
  v16h b;
#pragma unroll
  for (int j = 0; j < 8; ++j) { b[j] = lo[j]; b[j + 8] = hi[j]; }
  return b;
}

// ---------------------------------------------------------------------------
// Kernels
// ---------------------------------------------------------------------------
__global__ void f32_to_f16_kernel(const float* __restrict__ src, half_t* __restrict__ dst, int n) {
  int i = blockIdx.x * blockDim.x + threadIdx.x;
  if (i < n) dst[i] = (half_t)src[i];
}

// src: K x N f32 row-major; dst: N x K f16 row-major (transposed weight)
__global__ void conv_transpose_kernel(const float* __restrict__ src, half_t* __restrict__ dst,
                                      int K, int N) {
  int i = blockIdx.x * blockDim.x + threadIdx.x;
  if (i >= K * N) return;
  int k = i / N, n = i - k * N;
  dst[(size_t)n * K + k] = (half_t)src[i];
}

// vt[((b*H+h)*HD + d)*N + node] = qkv[(b*N+node)*768 + 512 + h*32 + d]
__global__ void v_transpose_kernel(const half_t* __restrict__ qkv, half_t* __restrict__ vt) {
  int i = blockIdx.x * blockDim.x + threadIdx.x;  // over B*N*D
  if (i >= B_ * N_ * D_) return;
  int d    = i & (D_ - 1);
  int node = (i >> 8) & (N_ - 1);
  int b    = i >> 17;
  int h    = d >> 5;
  int dd   = d & 31;
  vt[(((size_t)b * H_ + h) * HD_ + dd) * N_ + node] =
      qkv[(size_t)(b * N_ + node) * 768 + 512 + d];
}

__global__ void bias_scatter_kernel(const int* __restrict__ eidx, const int* __restrict__ etype,
                                    const float* __restrict__ ew, const float* __restrict__ et_emb,
                                    float* __restrict__ bias) {
  int i = blockIdx.x * blockDim.x + threadIdx.x;
  if (i >= B_ * E_) return;
  int b = i / E_, e = i - b * E_;
  int s = eidx[(size_t)b * 2 * E_ + e];
  int d = eidx[(size_t)b * 2 * E_ + E_ + e];
  int t = etype[(size_t)b * E_ + e];
  float w = (t == 2) ? ew[(size_t)b * E_ + e] : 0.0f;
#pragma unroll
  for (int h = 0; h < H_; ++h) {
    atomicAdd(&bias[(((size_t)b * H_ + h) * N_ + s) * N_ + d], et_emb[t * H_ + h] + w);
  }
}

// C[M,N] = act(A[M,K] @ B[K,N] + bias[N]) with B given TRANSPOSED (Bt: N x K f16).
// One wave computes a 16x64 tile (4 WMMAs share one A fragment per K-step).
__global__ __launch_bounds__(128)
void wmma_gemm_kernel(const half_t* __restrict__ A, const half_t* __restrict__ Bt,
                      const float* __restrict__ bias,
                      float* __restrict__ Cf, half_t* __restrict__ Ch,
                      int M, int N, int K, int act) {
  int lane  = threadIdx.x & 31;
  int wave  = threadIdx.x >> 5;
  int col0  = (blockIdx.x * 4 + wave) * 64;
  int row0  = blockIdx.y * 16;
  if (col0 >= N || row0 >= M) return;

  v8f acc0 = {}, acc1 = {}, acc2 = {}, acc3 = {};
  for (int k0 = 0; k0 < K; k0 += 32) {
    v16h a  = load_a_frag(A, K, row0, k0, lane);
    v16h b0 = load_b_frag_transposed(Bt, K, col0,      k0, lane);
    v16h b1 = load_b_frag_transposed(Bt, K, col0 + 16, k0, lane);
    v16h b2 = load_b_frag_transposed(Bt, K, col0 + 32, k0, lane);
    v16h b3 = load_b_frag_transposed(Bt, K, col0 + 48, k0, lane);
    acc0 = wmma_f16(a, b0, acc0);
    acc1 = wmma_f16(a, b1, acc1);
    acc2 = wmma_f16(a, b2, acc2);
    acc3 = wmma_f16(a, b3, acc3);
  }

  int n     = lane & 15;
  int mbase = (lane >> 4) * 8;
  v8f accs[4] = {acc0, acc1, acc2, acc3};
#pragma unroll
  for (int q = 0; q < 4; ++q) {
    int col = col0 + q * 16 + n;
    float bv = bias ? bias[col] : 0.0f;
#pragma unroll
    for (int r = 0; r < 8; ++r) {
      int row = row0 + mbase + r;
      float v = accs[q][r] + bv;
      if (act) v = gelu_exact(v);
      if (Cf) Cf[(size_t)row * N + col] = v;
      if (Ch) Ch[(size_t)row * N + col] = (half_t)v;
    }
  }
}

// One wave per (b, h, 16-row tile): scores = QK^T/sqrt(hd) + bias, mask, softmax, @V.
// qkv: (B*N, 768) f16 with Q at col h*32, K at 256+h*32.  vt: (B,H,HD,N) f16.
__global__ __launch_bounds__(32)
void attention_kernel(const half_t* __restrict__ qkv, const half_t* __restrict__ vt,
                      const float* __restrict__ bias, const int* __restrict__ node_mask,
                      half_t* __restrict__ attn_out) {
  __shared__ float  sc[16 * N_];   // 32 KB: f32 scores
  __shared__ half_t sp[16 * N_];   // 16 KB: f16 normalized probs
  int lane = threadIdx.x;
  int rt = blockIdx.x, h = blockIdx.y, b = blockIdx.z;
  int row0  = rt * 16;
  int n     = lane & 15;
  int mbase = (lane >> 4) * 8;

  const half_t* qbase  = qkv + (size_t)b * N_ * 768 + h * HD_;
  const half_t* kbase  = qkv + (size_t)b * N_ * 768 + 256 + h * HD_;
  const half_t* vtbase = vt + ((size_t)b * H_ + h) * HD_ * N_;
  const float*  bb     = bias + (((size_t)b * H_ + h) * N_ + row0) * N_;
  const float   scale  = 0.17677669529663687f;  // 1/sqrt(32)

  v16h aq = load_a_frag(qbase, 768, row0, 0, lane);

  // scores: Q(16x32) . K^T(32x512) in 32 column tiles (K rows are contiguous loads)
  for (int ct = 0; ct < N_ / 16; ++ct) {
    v16h bk = load_b_frag_transposed(kbase, 768, ct * 16, 0, lane);
    v8f c = {};
    c = wmma_f16(aq, bk, c);
    int col = ct * 16 + n;
    bool keyok = node_mask[b * N_ + col] != 0;
#pragma unroll
    for (int r = 0; r < 8; ++r) {
      float s = c[r] * scale + bb[(size_t)(mbase + r) * N_ + col];
      if (!keyok) s = -1e30f;
      sc[(mbase + r) * N_ + col] = s;
    }
  }
  __syncthreads();

  // softmax: lanes 0..15 each own one row; write normalized probs as f16
  if (lane < 16) {
    float mx = -1e30f;
    for (int j = 0; j < N_; ++j) mx = fmaxf(mx, sc[lane * N_ + j]);
    float sum = 0.0f;
    for (int j = 0; j < N_; ++j) {
      float e = __expf(sc[lane * N_ + j] - mx);
      sc[lane * N_ + j] = e;
      sum += e;
    }
    float inv = 1.0f / sum;
    for (int j = 0; j < N_; ++j) sp[lane * N_ + j] = (half_t)(sc[lane * N_ + j] * inv);
  }
  __syncthreads();

  // out(16x32) = probs(16x512) @ V(512x32); V supplied transposed (contiguous K loads)
  v8f acc0 = {}, acc1 = {};
  for (int kt = 0; kt < N_ / 32; ++kt) {
    v16h ap  = load_a_frag_lds(sp, kt * 32, lane);
    v16h bv0 = load_b_frag_transposed(vtbase, N_, 0,  kt * 32, lane);
    v16h bv1 = load_b_frag_transposed(vtbase, N_, 16, kt * 32, lane);
    acc0 = wmma_f16(ap, bv0, acc0);
    acc1 = wmma_f16(ap, bv1, acc1);
  }

  half_t* obase = attn_out + ((size_t)b * N_ + row0) * D_ + h * HD_;
#pragma unroll
  for (int r = 0; r < 8; ++r) {
    obase[(size_t)(mbase + r) * D_ + n]      = (half_t)acc0[r];
    obase[(size_t)(mbase + r) * D_ + 16 + n] = (half_t)acc1[r];
  }
}

// x = LayerNorm(x + y)*g + b ; also write f16 copy. One 256-thread block per row.
__global__ __launch_bounds__(256)
void resln_kernel(float* __restrict__ x, const float* __restrict__ y,
                  const float* __restrict__ g, const float* __restrict__ bta,
                  half_t* __restrict__ xh) {
  __shared__ float red[256];
  int row = blockIdx.x, t = threadIdx.x;
  float v = x[(size_t)row * D_ + t] + y[(size_t)row * D_ + t];
  red[t] = v;
  __syncthreads();
  for (int s = 128; s > 0; s >>= 1) { if (t < s) red[t] += red[t + s]; __syncthreads(); }
  float mu = red[0] * (1.0f / D_);
  __syncthreads();
  float dv = v - mu;
  red[t] = dv * dv;
  __syncthreads();
  for (int s = 128; s > 0; s >>= 1) { if (t < s) red[t] += red[t + s]; __syncthreads(); }
  float var = red[0] * (1.0f / D_);
  float o = dv * rsqrtf(var + 1e-5f) * g[t] + bta[t];
  x[(size_t)row * D_ + t]  = o;
  xh[(size_t)row * D_ + t] = (half_t)o;
}

// Pooling + conflict + 769->512->256->1 MLP head; one block per batch element.
__global__ __launch_bounds__(256)
void head_kernel(const float* __restrict__ x, const int* __restrict__ tmask,
                 const int* __restrict__ imask, const int* __restrict__ gidx,
                 const float* __restrict__ Wm1, const float* __restrict__ bm1,
                 const float* __restrict__ Wm2, const float* __restrict__ bm2,
                 const float* __restrict__ Wm3, const float* __restrict__ bm3,
                 float* __restrict__ out) {
  __shared__ float tp[D_], ip[D_], ge[D_], h1[512], h2[256];
  __shared__ float sconf;
  int b = blockIdx.x, t = threadIdx.x;
  float tpv = 0.0f, ipv = 0.0f;
  int tcnt = 0, icnt = 0;
  for (int nn = 0; nn < N_; ++nn) {
    float xv = x[((size_t)b * N_ + nn) * D_ + t];
    int tm = tmask[b * N_ + nn], im = imask[b * N_ + nn];
    if (tm) { tpv += xv; }
    if (im) { ipv += xv; }
    tcnt += (tm != 0);
    icnt += (im != 0);
  }
  float tc = fmaxf((float)tcnt, 1.0f), ic = fmaxf((float)icnt, 1.0f);
  tp[t] = tpv / tc;
  ip[t] = ipv / ic;
  ge[t] = x[((size_t)b * N_ + gidx[b]) * D_ + t];
  __syncthreads();
  if (t == 0) {
    float d12 = 0.f, n1 = 0.f, n2 = 0.f;
    for (int j = 0; j < D_; ++j) { d12 += tp[j] * ip[j]; n1 += tp[j] * tp[j]; n2 += ip[j] * ip[j]; }
    n1 = fmaxf(sqrtf(n1), 1e-6f);
    n2 = fmaxf(sqrtf(n2), 1e-6f);
    sconf = 1.0f - d12 / (n1 * n2);
  }
  __syncthreads();
  float conf = sconf;
  for (int c = t; c < 512; c += 256) {
    float acc = bm1[c];
    for (int j = 0; j < 256; ++j) acc += ge[j] * Wm1[(size_t)j * 512 + c];
    for (int j = 0; j < 256; ++j) acc += tp[j] * Wm1[(size_t)(256 + j) * 512 + c];
    for (int j = 0; j < 256; ++j) acc += ip[j] * Wm1[(size_t)(512 + j) * 512 + c];
    acc += conf * Wm1[(size_t)768 * 512 + c];
    h1[c] = gelu_exact(acc);
  }
  __syncthreads();
  {
    float acc = bm2[t];
    for (int j = 0; j < 512; ++j) acc += h1[j] * Wm2[(size_t)j * 256 + t];
    h2[t] = gelu_exact(acc);
  }
  __syncthreads();
  if (t == 0) {
    float acc = bm3[0];
    for (int j = 0; j < 256; ++j) acc += h2[j] * Wm3[j];
    out[b] = acc;
  }
}

// ---------------------------------------------------------------------------
// Host launcher
// ---------------------------------------------------------------------------
extern "C" void kernel_launch(void* const* d_in, const int* in_sizes, int n_in,
                              void* d_out, int out_size, void* d_ws, size_t ws_size,
                              hipStream_t stream) {
  (void)in_sizes; (void)n_in; (void)out_size; (void)ws_size;

  const float* node_feats = (const float*)d_in[0];
  const int*   node_mask  = (const int*)d_in[1];
  const int*   text_mask  = (const int*)d_in[2];
  const int*   image_mask = (const int*)d_in[3];
  const int*   gidx       = (const int*)d_in[4];
  const int*   eidx       = (const int*)d_in[5];
  const int*   etype      = (const int*)d_in[6];
  const float* ew         = (const float*)d_in[7];
  const float* W_in       = (const float*)d_in[8];
  const float* b_in       = (const float*)d_in[9];
  const float* Wqkv       = (const float*)d_in[10];
  const float* bqkv       = (const float*)d_in[11];
  const float* Wo         = (const float*)d_in[12];
  const float* bo         = (const float*)d_in[13];
  const float* ln1_g      = (const float*)d_in[14];
  const float* ln1_b      = (const float*)d_in[15];
  const float* ln2_g      = (const float*)d_in[16];
  const float* ln2_b      = (const float*)d_in[17];
  const float* Wff1       = (const float*)d_in[18];
  const float* bff1       = (const float*)d_in[19];
  const float* Wff2       = (const float*)d_in[20];
  const float* bff2       = (const float*)d_in[21];
  const float* et_emb     = (const float*)d_in[22];
  const float* Wm1        = (const float*)d_in[23];
  const float* bm1        = (const float*)d_in[24];
  const float* Wm2        = (const float*)d_in[25];
  const float* bm2        = (const float*)d_in[26];
  const float* Wm3        = (const float*)d_in[27];
  const float* bm3        = (const float*)d_in[28];

  // Bump allocator over workspace (256-byte aligned chunks)
  char* wsb = (char*)d_ws;
  size_t off = 0;
  auto balloc = [&](size_t bytes) -> void* {
    void* p = wsb + off;
    off = (off + bytes + 255) & ~(size_t)255;
    return p;
  };
  float*  bias   = (float*)balloc((size_t)B_ * H_ * N_ * N_ * 4);  // 134 MB
  float*  xf     = (float*)balloc((size_t)B_ * N_ * D_ * 4);
  float*  yf     = (float*)balloc((size_t)B_ * N_ * D_ * 4);
  half_t* xh     = (half_t*)balloc((size_t)B_ * N_ * D_ * 2);
  half_t* inh    = (half_t*)balloc((size_t)B_ * N_ * IN_ * 2);
  half_t* qkvh   = (half_t*)balloc((size_t)B_ * N_ * 768 * 2);
  half_t* vth    = (half_t*)balloc((size_t)B_ * H_ * HD_ * N_ * 2);
  half_t* atth   = (half_t*)balloc((size_t)B_ * N_ * D_ * 2);
  half_t* h1h    = (half_t*)balloc((size_t)B_ * N_ * FF_ * 2);
  half_t* Wint   = (half_t*)balloc((size_t)IN_ * D_ * 2);
  half_t* Wqkvt  = (half_t*)balloc((size_t)L_ * D_ * 768 * 2);
  half_t* Wot    = (half_t*)balloc((size_t)L_ * D_ * D_ * 2);
  half_t* Wff1t  = (half_t*)balloc((size_t)L_ * D_ * FF_ * 2);
  half_t* Wff2t  = (half_t*)balloc((size_t)L_ * FF_ * D_ * 2);

  const int M = B_ * N_;  // 8192 rows

  auto conv = [&](const float* s, half_t* d, size_t n) {
    f32_to_f16_kernel<<<dim3((unsigned)((n + 255) / 256)), dim3(256), 0, stream>>>(s, d, (int)n);
  };
  auto convT = [&](const float* s, half_t* d, int K, int Nn) {
    size_t n = (size_t)K * Nn;
    conv_transpose_kernel<<<dim3((unsigned)((n + 255) / 256)), dim3(256), 0, stream>>>(s, d, K, Nn);
  };
  auto gemm = [&](const half_t* A, const half_t* Bt, const float* bv,
                  float* Cf, half_t* Ch, int Nn, int K, int act) {
    wmma_gemm_kernel<<<dim3(Nn / 256, M / 16), dim3(128), 0, stream>>>(A, Bt, bv, Cf, Ch, M, Nn, K, act);
  };

  // 0) attention bias: zero + scatter-add over edges
  hipMemsetAsync(bias, 0, (size_t)B_ * H_ * N_ * N_ * 4, stream);
  bias_scatter_kernel<<<dim3((B_ * E_) / 256), dim3(256), 0, stream>>>(eidx, etype, ew, et_emb, bias);

  // 1) f32 -> f16 conversions; weights transposed to (N x K) for vectorized B-fragments
  conv(node_feats, inh, (size_t)B_ * N_ * IN_);
  convT(W_in, Wint, IN_, D_);
  for (int l = 0; l < L_; ++l) {
    convT(Wqkv + (size_t)l * D_ * 768, Wqkvt + (size_t)l * D_ * 768, D_, 768);
    convT(Wo   + (size_t)l * D_ * D_,  Wot   + (size_t)l * D_ * D_,  D_, D_);
    convT(Wff1 + (size_t)l * D_ * FF_, Wff1t + (size_t)l * D_ * FF_, D_, FF_);
    convT(Wff2 + (size_t)l * FF_ * D_, Wff2t + (size_t)l * FF_ * D_, FF_, D_);
  }

  // 2) input projection: x = node_feats @ W_in + b_in
  gemm(inh, Wint, b_in, xf, xh, D_, IN_, 0);

  // 3) transformer layers
  for (int l = 0; l < L_; ++l) {
    gemm(xh, Wqkvt + (size_t)l * D_ * 768, bqkv + (size_t)l * 768, nullptr, qkvh, 768, D_, 0);
    v_transpose_kernel<<<dim3((B_ * N_ * D_) / 256), dim3(256), 0, stream>>>(qkvh, vth);
    attention_kernel<<<dim3(N_ / 16, H_, B_), dim3(32), 0, stream>>>(qkvh, vth, bias, node_mask, atth);
    gemm(atth, Wot + (size_t)l * D_ * D_, bo + (size_t)l * D_, yf, nullptr, D_, D_, 0);
    resln_kernel<<<dim3(M), dim3(256), 0, stream>>>(xf, yf, ln1_g + (size_t)l * D_, ln1_b + (size_t)l * D_, xh);
    gemm(xh, Wff1t + (size_t)l * D_ * FF_, bff1 + (size_t)l * FF_, nullptr, h1h, FF_, D_, 1);
    gemm(h1h, Wff2t + (size_t)l * FF_ * D_, bff2 + (size_t)l * D_, yf, nullptr, D_, FF_, 0);
    resln_kernel<<<dim3(M), dim3(256), 0, stream>>>(xf, yf, ln2_g + (size_t)l * D_, ln2_b + (size_t)l * D_, xh);
  }

  // 4) pooling + conflict + MLP head -> logits (16,) f32
  head_kernel<<<dim3(B_), dim3(256), 0, stream>>>(xf, text_mask, image_mask, gidx,
                                                  Wm1, bm1, Wm2, bm2, Wm3, bm3,
                                                  (float*)d_out);
}
